// EdgeEmbed_32847909879961
// MI455X (gfx1250) — matrix operand order repaired and checked by reading, not compiled, not run on hardware
//
#include <hip/hip_runtime.h>
#include <stdint.h>

#define NODE_DIM 128
#define EDGE_DIM 128
#define N_RADIAL 16
#define N_EDGES  800000
#define K_REAL   272          // 2*NODE_DIM + N_RADIAL (rbf folded into weights)
#define K_PAD    288          // 9 * 32  (WMMA K-step = 32 for bf16)
#define M_TILE   128          // edges per workgroup
#define LDK      296          // padded LDS row stride (bf16 units); 296*2=592=37*16

typedef __attribute__((ext_vector_type(16))) __bf16 v16bf;
typedef __attribute__((ext_vector_type(8)))  float  v8f;
typedef __attribute__((ext_vector_type(4)))  int    v4i;
typedef __attribute__((address_space(1))) v4i gv4i;   // global v4i
typedef __attribute__((address_space(3))) v4i lv4i;   // LDS v4i

#if defined(__gfx1250__) && __has_builtin(__builtin_amdgcn_global_load_async_to_lds_b128) && \
    __has_builtin(__builtin_amdgcn_s_wait_asynccnt)
#define EE_ASYNC_LDS 1
#else
#define EE_ASYNC_LDS 0
#endif

__device__ __forceinline__ uint16_t f32_to_bf16(float f) {
    union { float f; uint32_t u; } c; c.f = f;
    uint32_t u = c.u;
    return (uint16_t)((u + 0x7FFFu + ((u >> 16) & 1u)) >> 16);   // RNE
}

// Convert 8 consecutive f32 -> 8 bf16 and store as one 16B chunk.
__device__ __forceinline__ void cvt8_store(uint16_t* dst, const float4* src) {
    float4 a = src[0], b = src[1];
    uint16_t t[8];
    t[0]=f32_to_bf16(a.x); t[1]=f32_to_bf16(a.y); t[2]=f32_to_bf16(a.z); t[3]=f32_to_bf16(a.w);
    t[4]=f32_to_bf16(b.x); t[5]=f32_to_bf16(b.y); t[6]=f32_to_bf16(b.z); t[7]=f32_to_bf16(b.w);
    uint4 u; __builtin_memcpy(&u, t, 16);
    *reinterpret_cast<uint4*>(dst) = u;
}

// ---------------------------------------------------------------------------
// Prep: build combined transposed weight Wt[n][k], n<128, k<288 (bf16).
//   k <  256 : edge_w[k][n]
//   256..271 : (rbf_w @ edge_w[256:384])[k-256][n]   (rbf projection folded in)
//   272..287 : 0  (zero pad filling out the 9th WMMA K-step)
// ---------------------------------------------------------------------------
__global__ void ee_build_wt(const float* __restrict__ rbf_w,
                            const float* __restrict__ edge_w,
                            uint16_t* __restrict__ wt) {
    int gid = blockIdx.x * blockDim.x + threadIdx.x;
    if (gid >= EDGE_DIM * K_PAD) return;
    int n = gid / K_PAD;
    int k = gid % K_PAD;
    float v = 0.f;
    if (k < 2 * NODE_DIM) {
        v = edge_w[k * EDGE_DIM + n];
    } else if (k < K_REAL) {
        int r = k - 2 * NODE_DIM;
        float acc = 0.f;
        for (int c = 0; c < EDGE_DIM; ++c)
            acc += rbf_w[r * EDGE_DIM + c] * edge_w[(2 * NODE_DIM + c) * EDGE_DIM + n];
        v = acc;
    }
    wt[gid] = f32_to_bf16(v);
}

// ---------------------------------------------------------------------------
// Main: 128 edges per WG. LDS-staged bf16 A (gathered feats) and Wt,
// 8 waves each compute a 16x128 output slab via v_wmma_f32_16x16x32_bf16.
// Wt is staged with the CDNA5 async Global->LDS pipe (no VGPR round-trip).
// ---------------------------------------------------------------------------
__global__ __launch_bounds__(256) void ee_edge_mlp(
    const int*   __restrict__ z,
    const float* __restrict__ rbf,
    const int*   __restrict__ idx_i,
    const int*   __restrict__ idx_j,
    const float* __restrict__ embed_table,
    const uint16_t* __restrict__ wt,      // [128][288] bf16
    const float* __restrict__ bias,
    float*       __restrict__ out)
{
    __shared__ __align__(16) uint16_t ldsA[M_TILE  * LDK];   // feat tile, bf16
    __shared__ __align__(16) uint16_t ldsB[EDGE_DIM * LDK];  // Wt (N-major), bf16

    const int tid   = threadIdx.x;
    const int edge0 = blockIdx.x * M_TILE;

    // ---- stage Wt -> LDS: 128 rows x 36 uint4 chunks (4608 over 256 thr) ----
#if EE_ASYNC_LDS
    for (int c = tid; c < EDGE_DIM * (K_PAD / 8); c += 256) {
        int n = c / (K_PAD / 8);
        int q = c % (K_PAD / 8);
        __builtin_amdgcn_global_load_async_to_lds_b128(
            (gv4i*)(reinterpret_cast<const uint4*>(wt) + c),
            (lv4i*)&ldsB[n * LDK + q * 8],
            /*offset=*/0, /*cpol=*/0);
    }
#else
    for (int c = tid; c < EDGE_DIM * (K_PAD / 8); c += 256) {
        int n = c / (K_PAD / 8);
        int q = c % (K_PAD / 8);
        uint4 src = reinterpret_cast<const uint4*>(wt)[c];
        *reinterpret_cast<uint4*>(&ldsB[n * LDK + q * 8]) = src;
    }
#endif

    // ---- gather + convert A tile: 2 threads per edge (overlaps async Wt) ----
    {
        const int e  = tid >> 1;        // 0..127
        const int s  = tid & 1;         // 0: j-part, 1: i-part
        const int eg = edge0 + e;
        const int node = (s == 0) ? idx_j[eg] : idx_i[eg];
        const int zz   = z[node];
        const float4* src = reinterpret_cast<const float4*>(embed_table + (size_t)zz * NODE_DIM);
        uint16_t* dst = &ldsA[e * LDK + s * NODE_DIM];
        #pragma unroll
        for (int q = 0; q < NODE_DIM / 8; ++q)          // 16 x (8 f32 -> 8 bf16)
            cvt8_store(dst + q * 8, src + 2 * q);
        // rbf: 8 f32 each thread
        const float4* rsrc = reinterpret_cast<const float4*>(rbf + (size_t)eg * N_RADIAL + s * 8);
        cvt8_store(&ldsA[e * LDK + 2 * NODE_DIM + s * 8], rsrc);
        if (s) {                                        // zero the K pad 272..287
            uint4 z4 = make_uint4(0u, 0u, 0u, 0u);
            uint4* pz = reinterpret_cast<uint4*>(&ldsA[e * LDK + K_REAL]);
            pz[0] = z4; pz[1] = z4;
        }
    }
#if EE_ASYNC_LDS
    __builtin_amdgcn_s_wait_asynccnt(0);   // retire this wave's async Wt loads
#endif
    __syncthreads();

    // ---- WMMA main loop ----
    const int lane = tid & 31;
    const int w    = tid >> 5;       // wave id -> M subtile
    const int half = lane >> 4;      // lane group (K/row split per ISA layout)
    const int nlo  = lane & 15;
    const int mrow = w * 16 + nlo;   // A row held by this lane (both halves)

    v8f acc[8];
    #pragma unroll
    for (int t = 0; t < 8; ++t) acc[t] = (v8f)(0.f);

    #pragma unroll
    for (int kk = 0; kk < K_PAD / 32; ++kk) {
        // A fragment: lane half h reads K = kk*32 + {8h..8h+7, 16+8h..16+8h+7}
        union { v16bf v; uint4 q[2]; } af;
        const uint16_t* abase = &ldsA[mrow * LDK + kk * 32 + half * 8];
        af.q[0] = *reinterpret_cast<const uint4*>(abase);
        af.q[1] = *reinterpret_cast<const uint4*>(abase + 16);

        #pragma unroll
        for (int t = 0; t < 8; ++t) {
            // B fragment: lane = col (t*16+nlo); K = kk*32 + {8h..8h+7, 16+8h..16+8h+7}
            union { v16bf v; uint4 q[2]; } bf_;
            const uint16_t* bbase = &ldsB[(t * 16 + nlo) * LDK + kk * 32 + half * 8];
            bf_.q[0] = *reinterpret_cast<const uint4*>(bbase);
            bf_.q[1] = *reinterpret_cast<const uint4*>(bbase + 16);
            acc[t] = __builtin_amdgcn_wmma_f32_16x16x32_bf16(
                false, af.v, false, bf_.v, (short)0, acc[t], false, false);
        }
    }

    // ---- epilogue: bias + swish, f32 stores ----
    const int mglob = edge0 + w * 16 + half * 8;   // + r  (C/D row layout)
    #pragma unroll
    for (int t = 0; t < 8; ++t) {
        const int col = t * 16 + nlo;
        const float b = bias[col];
        #pragma unroll
        for (int r = 0; r < 8; ++r) {
            float h  = acc[t][r] + b;
            float sw = h * (1.0f / (1.0f + __expf(-h)));
            out[(size_t)(mglob + r) * EDGE_DIM + col] = sw;
        }
    }
}

// ---------------------------------------------------------------------------
extern "C" void kernel_launch(void* const* d_in, const int* in_sizes, int n_in,
                              void* d_out, int out_size, void* d_ws, size_t ws_size,
                              hipStream_t stream) {
    const int*   z           = (const int*)  d_in[0];
    const float* rbf         = (const float*)d_in[1];
    const int*   idx_i       = (const int*)  d_in[2];
    const int*   idx_j       = (const int*)  d_in[3];
    const float* embed_table = (const float*)d_in[4];
    const float* rbf_w       = (const float*)d_in[5];
    const float* edge_w      = (const float*)d_in[6];
    const float* edge_b      = (const float*)d_in[7];
    float* out = (float*)d_out;

    uint16_t* wt = (uint16_t*)d_ws;   // 128*288 bf16 = 73,728 B

    ee_build_wt<<<(EDGE_DIM * K_PAD + 255) / 256, 256, 0, stream>>>(rbf_w, edge_w, wt);
    ee_edge_mlp<<<N_EDGES / M_TILE, 256, 0, stream>>>(
        z, rbf, idx_i, idx_j, embed_table, wt, edge_b, out);
}
